// Model_33913061769594
// MI455X (gfx1250) — matrix operand order repaired and checked by reading, not compiled, not run on hardware
//
#include <hip/hip_runtime.h>

typedef __attribute__((ext_vector_type(16))) _Float16 v16h;
typedef __attribute__((ext_vector_type(8)))  float    v8f;

#define G_TOT   1024
#define NPTS    (G_TOT * 25)
#define DX_F    0.0625f
#define INV2PI  0.15915494309189535f

__device__ __forceinline__ float fast_rcp(float x) {
  return __builtin_amdgcn_rcpf(x);
}

// ---------------------------------------------------------------------------
// Kernel 1: splat field.  field[p] = sum_j opac[j]*isd[j]*exp(power)*u[j]
// 25600 points, 1024 gaussians; gaussian params staged via LDS in 256-chunks.
// ---------------------------------------------------------------------------
__global__ __launch_bounds__(256) void field_kernel(
    const float* __restrict__ u, const float* __restrict__ means,
    const float* __restrict__ scaling, const float* __restrict__ transform,
    const float* __restrict__ opac, float* __restrict__ field) {
  __shared__ float smx[256], smy[256], sc0[256], sc1[256], sc2[256], spu[256];
  const int tid = threadIdx.x;
  const int p   = blockIdx.x * 256 + tid;          // grid = 100 blocks, exact
  const int g   = p / 25;
  const int k   = p % 25;
  const float ptx = means[2 * g + 0] + (float)((k % 5) - 2) * DX_F;
  const float pty = means[2 * g + 1] + (float)((k / 5) - 2) * DX_F;
  float s = 0.0f;
  for (int base = 0; base < G_TOT; base += 256) {
    const int j = base + tid;
    const float a = scaling[2 * j + 0];
    const float c = scaling[2 * j + 1];
    const float t = transform[j];
    const float cov00 = a * a;
    const float cov01 = a * t;
    const float cov11 = t * t + c * c;
    const float det   = cov00 * cov11 - cov01 * cov01;
    const float inv   = fast_rcp(det);
    sc0[tid] = cov11 * inv;
    sc1[tid] = -cov01 * inv;
    sc2[tid] = cov00 * inv;
    spu[tid] = opac[j] * INV2PI * sqrtf(inv) * u[j];
    smx[tid] = means[2 * j + 0];
    smy[tid] = means[2 * j + 1];
    __syncthreads();
    #pragma unroll 8
    for (int q = 0; q < 256; ++q) {
      const float ddx = ptx - smx[q];
      const float ddy = pty - smy[q];
      const float pw  = sc0[q] * ddx * ddx + 2.0f * sc1[q] * ddx * ddy +
                        sc2[q] * ddy * ddy;
      s += spu[q] * __expf(-0.5f * pw);
    }
    __syncthreads();
  }
  field[p] = s;
}

// ---------------------------------------------------------------------------
// Kernel 2: MLP heads via V_WMMA_F32_16X16X32_F16 (f16 in / f32 accumulate).
// grid.x = 64 row-tiles of 16 gaussians, 128 threads = 4 waves per block.
// All LDS data is staged in WMMA fragment-ready order so each A/B fragment
// is a single aligned v16h load (2x ds_load_b128) instead of 16 u16 gathers.
// ---------------------------------------------------------------------------
struct NetP {
  const float *conv_w, *conv_b, *pos_w, *pos_b;
  const float *l1_w, *l1_b, *l2_w, *l2_b, *l3_w, *l3_b;
  float* outp;
  int out_dim;
};
struct Net3 { NetP n[3]; };

// A-fragment element address for value at (row M, column/K index N) within an
// activation buffer laid out as [ks][lane(32)][elem(16)] halves.
// A-layout (contiguous k): lane = M + 16*((Kl>>4)&1), elem = Kl&15.
__device__ __forceinline__ int a_slot(int M, int N) {
  const int ks = N >> 5, Kl = N & 31;
  const int gp = (Kl >> 4) & 1;
  const int e  = Kl & 15;
  return ((ks << 5) + (gp << 4) + M) * 16 + e;
}

__device__ __forceinline__ v16h ldfragA(const _Float16* buf, int ks, int lane) {
  return *reinterpret_cast<const v16h*>(buf + (((ks << 5) + lane) << 4));
}
__device__ __forceinline__ v16h ldfragB(const _Float16* buf, int NT, int ks,
                                        int nt, int lane) {
  return *reinterpret_cast<const v16h*>(buf +
                                        (((ks * NT + nt) << 5) + lane) * 16);
}

template <int ACT>
__device__ __forceinline__ float activate(float z, float w1v, float w2v) {
  if (ACT == 0) return w1v * __sinf(z) + w2v * __cosf(z);      // wave act
  return z * fast_rcp(1.0f + __expf(-z));                      // fast silu
}

template <int ACT>
__device__ __forceinline__ void run_net(const float* __restrict__ field,
                                        const float* __restrict__ means,
                                        const NetP& p, const float* w1p,
                                        const float* w2p) {
  __shared__ __align__(32) _Float16 shW[128 * 128];  // frag-ready weights
  __shared__ __align__(32) _Float16 shX[4 * 32 * 16];  // activations ping
  __shared__ __align__(32) _Float16 shY[4 * 32 * 16];  // activations pong
  __shared__ float sh_m[32];

  const int tid  = threadIdx.x;
  const int lane = tid & 31;
  const int wave = tid >> 5;
  const int tile = blockIdx.x;

  float w1v = 0.0f, w2v = 0.0f;
  if (ACT == 0) { w1v = w1p[0]; w2v = w2p[0]; }

  // Warm GL2 for the big weight arrays while layer 0 runs.
  if (tid == 0) __builtin_prefetch((const void*)p.l1_w, 0, 1);
  if (tid == 1) __builtin_prefetch((const void*)p.l2_w, 0, 1);

  // ---- stage conv weights (K 25->32 pad, N=64, frag-ready) + img + means ---
  for (int idx = tid; idx < 32 * 64; idx += 128) {     // k = idx>>6, n = idx&63
    const int k = idx >> 6, n = idx & 63;
    const float v = (k < 25) ? p.conv_w[n * 25 + k] : 0.0f;
    const int g = k >> 4, e = k & 15;                  // B: K = 16*g + e
    shW[((((k >> 5) * 4 + (n >> 4)) << 5) + (g << 4) + (n & 15)) * 16 + e] =
        (_Float16)v;
  }
  for (int idx = tid; idx < 16 * 32; idx += 128) {     // img tile -> A-ready
    const int r = idx >> 5, c = idx & 31;
    const float v = (c < 25) ? field[(tile * 16 + r) * 25 + c] : 0.0f;
    shX[a_slot(r, c)] = (_Float16)v;
  }
  if (tid < 32) sh_m[tid] = means[tile * 32 + tid];
  __syncthreads();

  // ---- layer 0a: conv reduce  y(16x64) = act(img(16x32) @ Wc(32x64)+b) -----
  {
    v8f acc = {};
    const v16h a = ldfragA(shX, 0, lane);
    const v16h b = ldfragB(shW, 4, 0, wave, lane);
    acc = __builtin_amdgcn_wmma_f32_16x16x32_f16(false, a, false, b, (short)0,
                                                 acc, false, false);
    const int n = (wave << 4) + (lane & 15);
    #pragma unroll
    for (int r = 0; r < 8; ++r) {
      const int m = r + ((lane >> 4) << 3);
      shY[a_slot(m, n)] =
          (_Float16)activate<ACT>(acc[r] + p.conv_b[n], w1v, w2v);
    }
  }
  // ---- layer 0b: pos embed e(16x32) -> columns 64..95 of h ----------------
  for (int idx = tid; idx < 16 * 32; idx += 128) {
    const int r = idx >> 5, c = idx & 31;
    const float z = sh_m[2 * r] * p.pos_w[2 * c] +
                    sh_m[2 * r + 1] * p.pos_w[2 * c + 1] + p.pos_b[c];
    shY[a_slot(r, 64 + c)] = (_Float16)activate<ACT>(z, w1v, w2v);
  }
  // (columns 96..127 live in kstep region 3, which l1 never reads: K=96.)
  __syncthreads();

  // ---- stage l1 weights (K=96, N=128, frag-ready) -------------------------
  for (int idx = tid; idx < 96 * 128; idx += 128) {
    const int k = idx >> 7, n = idx & 127;
    const int g = (k >> 4) & 1, e = k & 15;
    shW[((((k >> 5) * 8 + (n >> 4)) << 5) + (g << 4) + (n & 15)) * 16 + e] =
        (_Float16)p.l1_w[n * 96 + k];
  }
  __syncthreads();

  // ---- l1: h1(16x128) = act(h(16x96) @ W1 + b) : 3 K-steps, 2 N-tiles/wave -
  {
    v8f acc0 = {}, acc1 = {};
    #pragma unroll
    for (int ks = 0; ks < 3; ++ks) {
      const v16h a  = ldfragA(shY, ks, lane);
      const v16h b0 = ldfragB(shW, 8, ks, wave, lane);
      acc0 = __builtin_amdgcn_wmma_f32_16x16x32_f16(false, a, false, b0,
                                                    (short)0, acc0, false, false);
      const v16h b1 = ldfragB(shW, 8, ks, wave + 4, lane);
      acc1 = __builtin_amdgcn_wmma_f32_16x16x32_f16(false, a, false, b1,
                                                    (short)0, acc1, false, false);
    }
    const int nb = lane & 15;
    #pragma unroll
    for (int r = 0; r < 8; ++r) {
      const int m  = r + ((lane >> 4) << 3);
      const int n0 = (wave << 4) + nb;
      const int n1 = ((wave + 4) << 4) + nb;
      shX[a_slot(m, n0)] =
          (_Float16)activate<ACT>(acc0[r] + p.l1_b[n0], w1v, w2v);
      shX[a_slot(m, n1)] =
          (_Float16)activate<ACT>(acc1[r] + p.l1_b[n1], w1v, w2v);
    }
  }
  __syncthreads();

  // ---- stage l2 weights (K=128, N=128, frag-ready) ------------------------
  for (int idx = tid; idx < 128 * 128; idx += 128) {
    const int k = idx >> 7, n = idx & 127;
    const int g = (k >> 4) & 1, e = k & 15;
    shW[((((k >> 5) * 8 + (n >> 4)) << 5) + (g << 4) + (n & 15)) * 16 + e] =
        (_Float16)p.l2_w[n * 128 + k];
  }
  __syncthreads();

  // ---- l2: h2(16x128) = act(h1(16x128) @ W2 + b) : 4 K-steps --------------
  {
    v8f acc0 = {}, acc1 = {};
    #pragma unroll
    for (int ks = 0; ks < 4; ++ks) {
      const v16h a  = ldfragA(shX, ks, lane);
      const v16h b0 = ldfragB(shW, 8, ks, wave, lane);
      acc0 = __builtin_amdgcn_wmma_f32_16x16x32_f16(false, a, false, b0,
                                                    (short)0, acc0, false, false);
      const v16h b1 = ldfragB(shW, 8, ks, wave + 4, lane);
      acc1 = __builtin_amdgcn_wmma_f32_16x16x32_f16(false, a, false, b1,
                                                    (short)0, acc1, false, false);
    }
    const int nb = lane & 15;
    #pragma unroll
    for (int r = 0; r < 8; ++r) {
      const int m  = r + ((lane >> 4) << 3);
      const int n0 = (wave << 4) + nb;
      const int n1 = ((wave + 4) << 4) + nb;
      shY[a_slot(m, n0)] =
          (_Float16)activate<ACT>(acc0[r] + p.l2_b[n0], w1v, w2v);
      shY[a_slot(m, n1)] =
          (_Float16)activate<ACT>(acc1[r] + p.l2_b[n1], w1v, w2v);
    }
  }
  __syncthreads();

  // ---- l3: out(16 x od) = h2 @ l3_w^T + b  (od in {1,2}: plain VALU) ------
  const int od = p.out_dim;
  if (tid < 16 * od) {
    const int row = tid / od, n = tid % od;
    float s = p.l3_b[n];
    #pragma unroll 8
    for (int k = 0; k < 128; ++k)
      s += (float)shY[a_slot(row, k)] * p.l3_w[n * 128 + k];
    p.outp[(tile * 16 + row) * od + n] = s;
  }
}

__global__ __launch_bounds__(128) void net_kernel_wave(
    const float* __restrict__ field, const float* __restrict__ means, NetP p,
    const float* __restrict__ w1p, const float* __restrict__ w2p) {
  run_net<0>(field, means, p, w1p, w2p);
}

__global__ __launch_bounds__(128) void net_kernel_silu(
    const float* __restrict__ field, const float* __restrict__ means, Net3 P3) {
  run_net<1>(field, means, P3.n[blockIdx.y], nullptr, nullptr);
}

// ---------------------------------------------------------------------------
// Kernel 3: assemble output (G, 6) = [u+d, means+dm, scaling*exp(ds), t+dt]
// ---------------------------------------------------------------------------
__global__ __launch_bounds__(256) void assemble_kernel(
    const float* __restrict__ u, const float* __restrict__ means,
    const float* __restrict__ scaling, const float* __restrict__ transform,
    const float* __restrict__ no0, const float* __restrict__ no1,
    const float* __restrict__ no2, const float* __restrict__ no3,
    float* __restrict__ out) {
  const int g = blockIdx.x * 256 + threadIdx.x;
  if (g >= G_TOT) return;
  out[g * 6 + 0] = u[g] + no0[g];
  out[g * 6 + 1] = means[2 * g + 0] + no1[2 * g + 0];
  out[g * 6 + 2] = means[2 * g + 1] + no1[2 * g + 1];
  out[g * 6 + 3] = scaling[2 * g + 0] * __expf(no2[2 * g + 0]);
  out[g * 6 + 4] = scaling[2 * g + 1] * __expf(no2[2 * g + 1]);
  out[g * 6 + 5] = transform[g] + no3[g];
}

// ---------------------------------------------------------------------------
extern "C" void kernel_launch(void* const* d_in, const int* in_sizes, int n_in,
                              void* d_out, int out_size, void* d_ws,
                              size_t ws_size, hipStream_t stream) {
  (void)n_in; (void)out_size; (void)ws_size;
  const float* u         = (const float*)d_in[0];
  const float* means     = (const float*)d_in[1];
  const float* scaling   = (const float*)d_in[2];
  const float* transform = (const float*)d_in[3];
  const float* opac      = (const float*)d_in[4];

  // Detect param-dict flattening convention:
  //   jax alphabetical: d_in[5] = model.conv_b (64)
  //   insertion order:  d_in[5] = model.conv_w (1600)
  const bool alpha = (in_sizes[5] == 64);
  int base[4], w1i, w2i;
  int ocw, ocb, opw, opb, ol1w, ol1b, ol2w, ol2b, ol3w, ol3b;
  if (alpha) {
    base[0] = 5; base[1] = 25; base[2] = 15; base[3] = 37;
    w1i = 35; w2i = 36;
    ocb = 0; ocw = 1; ol1b = 2; ol1w = 3; ol2b = 4; ol2w = 5;
    ol3b = 6; ol3w = 7; opb = 8; opw = 9;
  } else {
    base[0] = 5; base[1] = 15; base[2] = 25; base[3] = 35;
    w1i = 45; w2i = 46;
    ocw = 0; ocb = 1; opw = 2; opb = 3; ol1w = 4; ol1b = 5;
    ol2w = 6; ol2b = 7; ol3w = 8; ol3b = 9;
  }
  const int outd[4] = {1, 2, 2, 1};

  float* ws    = (float*)d_ws;
  float* field = ws;                       // 25600 f32
  float* no[4];
  no[0] = ws + NPTS;                       // 1024  (delta)
  no[1] = no[0] + 1024;                    // 2048  (translation)
  no[2] = no[1] + 2048;                    // 2048  (dscale)
  no[3] = no[2] + 2048;                    // 1024  (dtransform)

  NetP np[4];
  for (int i = 0; i < 4; ++i) {
    const int b = base[i];
    np[i].conv_w = (const float*)d_in[b + ocw];
    np[i].conv_b = (const float*)d_in[b + ocb];
    np[i].pos_w  = (const float*)d_in[b + opw];
    np[i].pos_b  = (const float*)d_in[b + opb];
    np[i].l1_w   = (const float*)d_in[b + ol1w];
    np[i].l1_b   = (const float*)d_in[b + ol1b];
    np[i].l2_w   = (const float*)d_in[b + ol2w];
    np[i].l2_b   = (const float*)d_in[b + ol2b];
    np[i].l3_w   = (const float*)d_in[b + ol3w];
    np[i].l3_b   = (const float*)d_in[b + ol3b];
    np[i].outp   = no[i];
    np[i].out_dim = outd[i];
  }
  const float* w1 = (const float*)d_in[w1i];
  const float* w2 = (const float*)d_in[w2i];

  field_kernel<<<NPTS / 256, 256, 0, stream>>>(u, means, scaling, transform,
                                               opac, field);
  net_kernel_wave<<<dim3(64, 1), 128, 0, stream>>>(field, means, np[0], w1, w2);
  Net3 P3;
  P3.n[0] = np[1]; P3.n[1] = np[2]; P3.n[2] = np[3];
  net_kernel_silu<<<dim3(64, 3), 128, 0, stream>>>(field, means, P3);
  assemble_kernel<<<(G_TOT + 255) / 256, 256, 0, stream>>>(
      u, means, scaling, transform, no[0], no[1], no[2], no[3], (float*)d_out);
}
